// ReverbAugment_66185446031577
// MI455X (gfx1250) — matrix operand order repaired and checked by reading, not compiled, not run on hardware
//
#include <hip/hip_runtime.h>

#define SAMPLES 160000
#define BATCH   32
#define KLEN    16000
#define PADL    16383            // chosen so (PADL - (KLEN-1)) % 8 == 0 -> 16B-aligned b128 loads
#define XS      176400           // padded row length (elements), multiple of 8
#define CHUNKS  501              // ceil((KLEN+15)/32) + edge

#define APRE_HALVES (CHUNKS * 512)   // per chunk: 2 runs x 32 lanes x 8 halves = 512 halves (1 KB)
#define WS_MAX_OFF  516096
#define WS_X_OFF    524288

typedef _Float16 v8h  __attribute__((ext_vector_type(8)));
typedef _Float16 v16h __attribute__((ext_vector_type(16)));
typedef float    v8f  __attribute__((ext_vector_type(8)));
typedef float    v4f  __attribute__((ext_vector_type(4)));

// ---------------- Prep 1: precompute Toeplitz filter operand in WMMA A-layout ----------------
// A(o=32c)[m,k] = rir[32c + k - m], stored per ISA 16-bit A 16x32 layout:
//   lane L: m = L&15, g = L>>4 ; run0 halves h=0..7 -> k = g*8+h ; run1 -> k = 16+g*8+h
__global__ void ReverbAugment_prep_a(const float* __restrict__ rir,
                                     _Float16* __restrict__ apre,
                                     unsigned int* __restrict__ maxslot) {
    int e = blockIdx.x * blockDim.x + threadIdx.x;
    if (e == 0) *maxslot = 0u;                 // re-init reduction slot every launch
    if (e >= APRE_HALVES) return;
    int c    = e >> 9;
    int rem  = e & 511;
    int j    = rem >> 8;                       // run 0/1
    int rem2 = rem & 255;
    int lane = rem2 >> 3;
    int h8   = rem2 & 7;
    int m = lane & 15, g = lane >> 4;
    int k = j ? (16 + g * 8 + h8) : (g * 8 + h8);
    int idx = 32 * c + k - m;
    _Float16 v = (_Float16)0.0f;
    if (idx >= 0 && idx < KLEN) v = (_Float16)rir[idx];
    apre[e] = v;                               // layout offset == e by construction
}

// ---------------- Prep 2: f32 waveform -> zero-padded f16 rows ----------------
__global__ void ReverbAugment_prep_x(const float* __restrict__ x,
                                     _Float16* __restrict__ xp) {
    long e = (long)blockIdx.x * blockDim.x + threadIdx.x;
    if (e >= (long)BATCH * XS) return;
    int b = (int)(e / XS);
    int i = (int)(e % XS);
    int s = i - PADL;
    float v = (s >= 0 && s < SAMPLES) ? x[(long)b * SAMPLES + s] : 0.0f;
    xp[e] = (_Float16)v;
}

// ---------------- Main: implicit-GEMM convolution on the WMMA pipe ----------------
// One wave per 16-sample time tile; each wave computes BOTH 16-batch groups with
// two accumulators, reusing one Toeplitz A load against two B loads (A is
// batch-independent). Runtime trip count keeps the loop compact (no full unroll).
__global__ __launch_bounds__(256, 1)
void ReverbAugment_conv_wmma(const _Float16* __restrict__ apre,
                             const _Float16* __restrict__ xp,
                             float* __restrict__ out, int nch) {
    const int lane = threadIdx.x & 31;
    const int w    = threadIdx.x >> 5;
    const int tile = blockIdx.x * 8 + w;
    const int t0   = tile * 16;
    const int n    = lane & 15;
    const int g    = lane >> 4;

    const _Float16* Ap  = apre + lane * 8;                                  // 16B-aligned
    const size_t    bofs = (size_t)(PADL - (KLEN - 1)) + t0 + g * 8;        // 16B-aligned
    const _Float16* Bp0 = xp + (size_t)n        * XS + bofs;                // batch group 0
    const _Float16* Bp1 = xp + (size_t)(16 + n) * XS + bofs;                // batch group 1

    v8f acc0 = {0.f, 0.f, 0.f, 0.f, 0.f, 0.f, 0.f, 0.f};
    v8f acc1 = {0.f, 0.f, 0.f, 0.f, 0.f, 0.f, 0.f, 0.f};

#pragma unroll 2
    for (int c = 0; c < nch; ++c) {
        // A: two 8-half runs in register layout (shared by both batch groups)
        v8h a0 = *(const v8h*)(Ap + (size_t)c * 512);
        v8h a1 = *(const v8h*)(Ap + (size_t)c * 512 + 256);
        // B: two 8-half K-runs per lane (k = g*8+h, 16+g*8+h), one per batch group
        v8h b00 = *(const v8h*)(Bp0 + 32 * c);
        v8h b01 = *(const v8h*)(Bp0 + 32 * c + 16);
        v8h b10 = *(const v8h*)(Bp1 + 32 * c);
        v8h b11 = *(const v8h*)(Bp1 + 32 * c + 16);
        __builtin_prefetch(Bp0 + 32 * (c + 12), 0, 3);   // near-cache prefetch of B streams
        __builtin_prefetch(Bp1 + 32 * (c + 12), 0, 3);

        v16h A  = __builtin_shufflevector(a0,  a1,  0,1,2,3,4,5,6,7,8,9,10,11,12,13,14,15);
        v16h B0 = __builtin_shufflevector(b00, b01, 0,1,2,3,4,5,6,7,8,9,10,11,12,13,14,15);
        v16h B1 = __builtin_shufflevector(b10, b11, 0,1,2,3,4,5,6,7,8,9,10,11,12,13,14,15);

        acc0 = __builtin_amdgcn_wmma_f32_16x16x32_f16(
            false, A, false, B0, (short)0, acc0, false, false);
        acc1 = __builtin_amdgcn_wmma_f32_16x16x32_f16(
            false, A, false, B1, (short)0, acc1, false, false);
    }

    // D layout: lane -> n = lane&15 ; element r -> m = g*8 + r  => 8 contiguous times per lane
    float* orow0 = out + (size_t)n        * SAMPLES + t0 + g * 8;
    float* orow1 = out + (size_t)(16 + n) * SAMPLES + t0 + g * 8;
    *(v4f*)orow0       = __builtin_shufflevector(acc0, acc0, 0, 1, 2, 3);
    *(v4f*)(orow0 + 4) = __builtin_shufflevector(acc0, acc0, 4, 5, 6, 7);
    *(v4f*)orow1       = __builtin_shufflevector(acc1, acc1, 0, 1, 2, 3);
    *(v4f*)(orow1 + 4) = __builtin_shufflevector(acc1, acc1, 4, 5, 6, 7);
}

// ---------------- Reduction: global max |y| (non-negative float bits compare as uint) ----------------
__global__ void ReverbAugment_absmax(const float* __restrict__ y,
                                     unsigned int* __restrict__ slot, int nelem) {
    float m = 0.0f;
    for (long i = (long)blockIdx.x * blockDim.x + threadIdx.x; i < nelem;
         i += (long)gridDim.x * blockDim.x)
        m = fmaxf(m, fabsf(y[i]));
    for (int off = 16; off > 0; off >>= 1)
        m = fmaxf(m, __shfl_xor(m, off, 32));
    if ((threadIdx.x & 31) == 0) atomicMax(slot, __float_as_uint(m));
}

__global__ void ReverbAugment_scale(float* __restrict__ y,
                                    const unsigned int* __restrict__ slot, int nelem) {
    float s = 1.0f / (__uint_as_float(*slot) + 1e-8f);
    long i = (long)blockIdx.x * blockDim.x + threadIdx.x;
    if (i < nelem) y[i] *= s;
}

extern "C" void kernel_launch(void* const* d_in, const int* in_sizes, int n_in,
                              void* d_out, int out_size, void* d_ws, size_t ws_size,
                              hipStream_t stream) {
    const float* wav = (const float*)d_in[0];   // [32, 160000] f32
    const float* rir = (const float*)d_in[1];   // [16000] f32
    float* out = (float*)d_out;                 // [32, 160000] f32

    char* ws = (char*)d_ws;                     // needs ~11.8 MB
    _Float16*     apre = (_Float16*)(ws);
    unsigned int* slot = (unsigned int*)(ws + WS_MAX_OFF);
    _Float16*     xp   = (_Float16*)(ws + WS_X_OFF);

    ReverbAugment_prep_a<<<(APRE_HALVES + 255) / 256, 256, 0, stream>>>(rir, apre, slot);
    ReverbAugment_prep_x<<<(BATCH * XS) / 256, 256, 0, stream>>>(wav, xp);   // 22050 blocks exact
    ReverbAugment_conv_wmma<<<SAMPLES / 16 / 8, 256, 0, stream>>>(apre, xp, out, CHUNKS);
    ReverbAugment_absmax<<<2048, 256, 0, stream>>>(out, slot, out_size);
    ReverbAugment_scale<<<(out_size + 255) / 256, 256, 0, stream>>>(out, slot, out_size);
}
